// CamadaDecoder_81363860455555
// MI455X (gfx1250) — matrix operand order repaired and checked by reading, not compiled, not run on hardware
//
#include <hip/hip_runtime.h>
#include <hip/hip_bf16.h>
#include <stdint.h>

// ---------------------------------------------------------------------------
// Types for CDNA5 WMMA (wave32): v_wmma_f32_16x16x32_bf16
// ---------------------------------------------------------------------------
typedef __attribute__((ext_vector_type(16))) __bf16 v16bf;
typedef __attribute__((ext_vector_type(8)))  float  v8f;

union BF16Frag {
    unsigned short u[16];
    v16bf          v;
};

__device__ __forceinline__ unsigned short f32_to_bf16(float f) {
    union { float f; uint32_t u; } c; c.f = f;
    uint32_t x = c.u;
    uint32_t r = x + 0x7FFFu + ((x >> 16) & 1u);   // round-to-nearest-even
    return (unsigned short)(r >> 16);
}

// Async global -> LDS copy of 16 bytes per lane (CDNA5, ASYNCcnt-tracked).
// Flat shared-pointer low 32 bits are the LDS byte address (ISA 10.2).
__device__ __forceinline__ void async_copy_b128(unsigned short* lds_dst,
                                                const unsigned short* g_src) {
    unsigned lo = (unsigned)(uintptr_t)lds_dst;
    asm volatile("global_load_async_to_lds_b128 %0, %1, off"
                 :: "v"(lo), "v"(g_src) : "memory");
}
__device__ __forceinline__ void async_wait0() {
    asm volatile("s_wait_asynccnt 0" ::: "memory");
}

// ---------------------------------------------------------------------------
// fp32 -> bf16 cast, 4 elements / thread
// ---------------------------------------------------------------------------
__global__ __launch_bounds__(256)
void cast_f32_bf16_kernel(const float* __restrict__ src,
                          unsigned short* __restrict__ dst, long long n4) {
    long long i = (long long)blockIdx.x * blockDim.x + threadIdx.x;
    long long stride = (long long)gridDim.x * blockDim.x;
    for (; i < n4; i += stride) {
        float4 f = ((const float4*)src)[i];
        ushort4 o;
        o.x = f32_to_bf16(f.x); o.y = f32_to_bf16(f.y);
        o.z = f32_to_bf16(f.z); o.w = f32_to_bf16(f.w);
        ((ushort4*)dst)[i] = o;
    }
}

// ---------------------------------------------------------------------------
// Tiled bf16 WMMA GEMM:  C[b] = epilogue( alpha * A[b] @ op(B[b]) + bias )
//   block tile 128x128, BK=32, 256 threads = 8 waves, wave tile 32x64.
//   Double-buffered LDS: async global->LDS copies for tile i+1 overlap the
//   WMMA stream for tile i; s_wait_asynccnt lands after the compute.
//   B is ALWAYS staged into LDS as [n][k] (K-contiguous) so both A and B
//   fragments are read with 2x ds_load_b128 each.
//   TRANS_B=true  : B is [N,K] row-major (Q @ K^T)  -> direct async copy
//   TRANS_B=false : B is [K,N] row-major (weights)  -> register transpose
//   OUT_BF16      : write bf16 (next GEMM input) else fp32
// Shapes are multiples of the tile sizes for every call here.
// ---------------------------------------------------------------------------
template<bool TRANS_B, bool OUT_BF16>
__global__ __launch_bounds__(256)
void gemm_bf16_wmma_kernel(const unsigned short* __restrict__ A,
                           const unsigned short* __restrict__ Bm,
                           void* __restrict__ Cout,
                           const float* __restrict__ bias,
                           int M, int N, int K,
                           long long batchA, long long batchB, long long batchC,
                           float alpha, int relu, int causal) {
    constexpr int BM = 128, BN = 128, BK = 32, PAD = 8, LDT = BK + PAD;

    __shared__ unsigned short As[2][BM][LDT];   // 2 x 10240 B, [m][k]
    __shared__ unsigned short Bs[2][BN][LDT];   // 2 x 10240 B, [n][k]

    const long long b = blockIdx.z;
    const unsigned short* Ab = A  + b * batchA;
    const unsigned short* Bb = Bm + b * batchB;

    const int tileM = blockIdx.y * BM;
    const int tileN = blockIdx.x * BN;

    const int tid  = threadIdx.x;
    const int lane = tid & 31;
    const int wave = tid >> 5;
    const int wm   = wave >> 1;        // 0..3 : 32-row band
    const int wn   = wave & 1;         // 0..1 : 64-col band
    const int lrow = lane & 15;
    const int lh   = lane >> 4;        // half-wave select

    // staging coordinates
    const int sr = (tid * 8) >> 5, sc = (tid * 8) & 31;          // iter 0
    const int sr1 = ((tid + 256) * 8) >> 5, sc1 = ((tid + 256) * 8) & 31;
    const int kc = (tid >> 5) * 4;     // NN-B transpose chunk: 4k x 4n
    const int nc = (tid & 31) * 4;

    auto stage_async_A = [&](int buf, int k0) {
        const unsigned short* g0 = Ab + (long long)(tileM + sr) * K + (k0 + sc);
        const unsigned short* g1 = Ab + (long long)(tileM + sr1) * K + (k0 + sc1);
        async_copy_b128(&As[buf][sr][sc], g0);
        async_copy_b128(&As[buf][sr1][sc1], g1);
        if (k0 + BK < K) __builtin_prefetch(g0 + BK, 0, 3);
    };
    auto stage_async_B = [&](int buf, int k0) {   // TRANS_B only
        const unsigned short* g0 = Bb + (long long)(tileN + sr) * K + (k0 + sc);
        const unsigned short* g1 = Bb + (long long)(tileN + sr1) * K + (k0 + sc1);
        async_copy_b128(&Bs[buf][sr][sc], g0);
        async_copy_b128(&Bs[buf][sr1][sc1], g1);
        if (k0 + BK < K) __builtin_prefetch(g0 + BK, 0, 3);
    };
    auto load_B_rows = [&](ushort4* rowv, int k0) {   // NN only
        #pragma unroll
        for (int j = 0; j < 4; ++j)
            rowv[j] = *(const ushort4*)(Bb + (long long)(k0 + kc + j) * N
                                           + (tileN + nc));
    };
    auto store_B_trans = [&](int buf, const ushort4* rowv) {
        #pragma unroll
        for (int t = 0; t < 4; ++t) {
            ushort4 col;
            col.x = ((const unsigned short*)&rowv[0])[t];
            col.y = ((const unsigned short*)&rowv[1])[t];
            col.z = ((const unsigned short*)&rowv[2])[t];
            col.w = ((const unsigned short*)&rowv[3])[t];
            *(ushort4*)&Bs[buf][nc + t][kc] = col;   // b64 store, K-contiguous
        }
    };

    v8f acc[2][4];
    #pragma unroll
    for (int i = 0; i < 2; ++i)
        #pragma unroll
        for (int j = 0; j < 4; ++j)
            #pragma unroll
            for (int e = 0; e < 8; ++e) acc[i][j][e] = 0.0f;

    // ---- prologue: stage tile 0 into buffer 0
    {
        ushort4 rowv0[4];
        if (!TRANS_B) load_B_rows(rowv0, 0);
        stage_async_A(0, 0);
        if (TRANS_B) stage_async_B(0, 0);
        else         store_B_trans(0, rowv0);
        async_wait0();
        __syncthreads();
    }

    int cur = 0;
    for (int k0 = 0; k0 < K; k0 += BK) {
        const int nxt = cur ^ 1;
        const bool more = (k0 + BK) < K;

        // ---- issue next tile's copies before computing (latency overlap)
        ushort4 rowv[4];
        if (!TRANS_B && more) load_B_rows(rowv, k0 + BK);
        if (more) {
            stage_async_A(nxt, k0 + BK);
            if (TRANS_B) stage_async_B(nxt, k0 + BK);
        }

        // ---- A fragments (16-bit A 16x32 layout): 2x ds_load_b128 each
        BF16Frag afr[2];
        #pragma unroll
        for (int mt = 0; mt < 2; ++mt) {
            const int r = wm * 32 + mt * 16 + lrow;
            #pragma unroll
            for (int e = 0; e < 16; ++e) {
                int kk = 16 * (e >> 3) + 8 * lh + (e & 7);
                afr[mt].u[e] = As[cur][r][kk];
            }
        }
        // ---- B fragments (16-bit B 32x16: same k order as A, K=8*half+e, +16 for e>=8)
        BF16Frag bfr[4];
        #pragma unroll
        for (int nt = 0; nt < 4; ++nt) {
            const int n = wn * 64 + nt * 16 + lrow;
            #pragma unroll
            for (int e = 0; e < 16; ++e)
                bfr[nt].u[e] = Bs[cur][n][16 * (e >> 3) + 8 * lh + (e & 7)];
        }
        // ---- 8 WMMAs per wave per K-step
        #pragma unroll
        for (int mt = 0; mt < 2; ++mt)
            #pragma unroll
            for (int nt = 0; nt < 4; ++nt)
                acc[mt][nt] = __builtin_amdgcn_wmma_f32_16x16x32_bf16(
                    false, afr[mt].v, false, bfr[nt].v,
                    (short)0, acc[mt][nt], false, false);

        // ---- finish staging next tile, publish, swap
        if (!TRANS_B && more) store_B_trans(nxt, rowv);
        async_wait0();
        __syncthreads();
        cur = nxt;
    }

    // ---- epilogue (C/D layout: m = vgpr + 8*half, n = lane&15)
    float* Cf = OUT_BF16 ? nullptr : ((float*)Cout + b * batchC);
    unsigned short* Cb = OUT_BF16 ? ((unsigned short*)Cout + b * batchC) : nullptr;

    #pragma unroll
    for (int mt = 0; mt < 2; ++mt) {
        #pragma unroll
        for (int nt = 0; nt < 4; ++nt) {
            const int gn = tileN + wn * 64 + nt * 16 + lrow;
            const float bv = bias ? bias[gn] : 0.0f;
            #pragma unroll
            for (int r = 0; r < 8; ++r) {
                const int gm = tileM + wm * 32 + mt * 16 + r + 8 * lh;
                float v = acc[mt][nt][r] * alpha + bv;
                if (relu) v = fmaxf(v, 0.0f);
                if (causal && gn > gm) v = -__builtin_inff();
                const long long idx = (long long)gm * N + gn;
                if (OUT_BF16) Cb[idx] = f32_to_bf16(v);
                else          Cf[idx] = v;
            }
        }
    }
}

// ---------------------------------------------------------------------------
// Row softmax: fp32 scores [rows, 2048] -> bf16 P. One block (256 thr) / row,
// 8 contiguous elements per thread (float4-vectorized).
// ---------------------------------------------------------------------------
__global__ __launch_bounds__(256)
void softmax_rows_kernel(const float* __restrict__ S,
                         unsigned short* __restrict__ P, int cols) {
    const long long row = blockIdx.x;
    const float* s = S + row * (long long)cols;
    unsigned short* p = P + row * (long long)cols;
    const int tid = threadIdx.x;

    __shared__ float red[256];

    float4 a = ((const float4*)s)[tid * 2];
    float4 bq = ((const float4*)s)[tid * 2 + 1];
    float ev[8] = {a.x, a.y, a.z, a.w, bq.x, bq.y, bq.z, bq.w};

    float vmax = -__builtin_inff();
    #pragma unroll
    for (int i = 0; i < 8; ++i) vmax = fmaxf(vmax, ev[i]);
    red[tid] = vmax; __syncthreads();
    for (int st = 128; st > 0; st >>= 1) {
        if (tid < st) red[tid] = fmaxf(red[tid], red[tid + st]);
        __syncthreads();
    }
    const float m = red[0]; __syncthreads();

    float sum = 0.0f;
    #pragma unroll
    for (int i = 0; i < 8; ++i) {
        ev[i] = __expf(ev[i] - m);             // exp(-inf - m) == 0 for masked
        sum += ev[i];
    }
    red[tid] = sum; __syncthreads();
    for (int st = 128; st > 0; st >>= 1) {
        if (tid < st) red[tid] += red[tid + st];
        __syncthreads();
    }
    const float inv = 1.0f / red[0];

    ushort4 o0, o1;
    o0.x = f32_to_bf16(ev[0] * inv); o0.y = f32_to_bf16(ev[1] * inv);
    o0.z = f32_to_bf16(ev[2] * inv); o0.w = f32_to_bf16(ev[3] * inv);
    o1.x = f32_to_bf16(ev[4] * inv); o1.y = f32_to_bf16(ev[5] * inv);
    o1.z = f32_to_bf16(ev[6] * inv); o1.w = f32_to_bf16(ev[7] * inv);
    ((ushort4*)p)[tid * 2]     = o0;
    ((ushort4*)p)[tid * 2 + 1] = o1;
}

// ---------------------------------------------------------------------------
// Fused residual add + LayerNorm. One block per row of D=1024, 4 contiguous
// elements per thread. Writes fp32 and optionally bf16.
// ---------------------------------------------------------------------------
__global__ __launch_bounds__(256)
void add_layernorm_kernel(const float* __restrict__ X,
                          const float* __restrict__ R,
                          const float* __restrict__ gamma,
                          const float* __restrict__ beta,
                          float* __restrict__ Yf,
                          unsigned short* __restrict__ Ybf, int D) {
    const long long row = blockIdx.x;
    const int tid = threadIdx.x;

    __shared__ float rsum[256];
    __shared__ float rsq[256];

    float4 xv = ((const float4*)(X + row * (long long)D))[tid];
    float4 rv = ((const float4*)(R + row * (long long)D))[tid];
    float v[4] = {xv.x + rv.x, xv.y + rv.y, xv.z + rv.z, xv.w + rv.w};

    float sum = 0.0f, sq = 0.0f;
    #pragma unroll
    for (int i = 0; i < 4; ++i) { sum += v[i]; sq += v[i] * v[i]; }
    rsum[tid] = sum; rsq[tid] = sq; __syncthreads();
    for (int st = 128; st > 0; st >>= 1) {
        if (tid < st) { rsum[tid] += rsum[tid + st]; rsq[tid] += rsq[tid + st]; }
        __syncthreads();
    }
    const float invD = 1.0f / (float)D;
    const float mu   = rsum[0] * invD;
    const float var  = rsq[0] * invD - mu * mu;
    const float rs   = rsqrtf(var + 1e-5f);

    float4 gv = ((const float4*)gamma)[tid];
    float4 bv = ((const float4*)beta)[tid];
    float4 o;
    o.x = (v[0] - mu) * rs * gv.x + bv.x;
    o.y = (v[1] - mu) * rs * gv.y + bv.y;
    o.z = (v[2] - mu) * rs * gv.z + bv.z;
    o.w = (v[3] - mu) * rs * gv.w + bv.w;
    ((float4*)(Yf + row * (long long)D))[tid] = o;
    if (Ybf) {
        ushort4 ob;
        ob.x = f32_to_bf16(o.x); ob.y = f32_to_bf16(o.y);
        ob.z = f32_to_bf16(o.z); ob.w = f32_to_bf16(o.w);
        ((ushort4*)(Ybf + row * (long long)D))[tid] = ob;
    }
}

// ---------------------------------------------------------------------------
// Orchestration
// ---------------------------------------------------------------------------
extern "C" void kernel_launch(void* const* d_in, const int* in_sizes, int n_in,
                              void* d_out, int out_size, void* d_ws, size_t ws_size,
                              hipStream_t stream) {
    (void)in_sizes; (void)n_in; (void)out_size; (void)ws_size;

    constexpr int B = 4, S = 2048, D = 1024, DFF = 4096;
    constexpr long long BS = (long long)B * S;          // 8192 rows

    const float* y    = (const float*)d_in[0];
    const float* Z    = (const float*)d_in[1];
    const float* WQ1  = (const float*)d_in[2];
    const float* WK1  = (const float*)d_in[3];
    const float* WV1  = (const float*)d_in[4];
    const float* WQ2  = (const float*)d_in[5];
    const float* WK2  = (const float*)d_in[6];
    const float* WV2  = (const float*)d_in[7];
    const float* Wff1 = (const float*)d_in[8];
    const float* bff1 = (const float*)d_in[9];
    const float* Wff2 = (const float*)d_in[10];
    const float* bff2 = (const float*)d_in[11];
    const float* g1   = (const float*)d_in[12];
    const float* be1  = (const float*)d_in[13];
    const float* g2   = (const float*)d_in[14];
    const float* be2  = (const float*)d_in[15];
    const float* g3   = (const float*)d_in[16];
    const float* be3  = (const float*)d_in[17];

    // ---- workspace layout (256B aligned regions) ----
    size_t off = 0;
    auto alloc = [&](size_t bytes) -> void* {
        off = (off + 255) & ~(size_t)255;
        void* p = (char*)d_ws + off;
        off += bytes;
        return p;
    };
    unsigned short* ybf   = (unsigned short*)alloc(BS * D * 2);
    unsigned short* zbf   = (unsigned short*)alloc(BS * D * 2);
    unsigned short* wq1b  = (unsigned short*)alloc((size_t)D * D * 2);
    unsigned short* wk1b  = (unsigned short*)alloc((size_t)D * D * 2);
    unsigned short* wv1b  = (unsigned short*)alloc((size_t)D * D * 2);
    unsigned short* wq2b  = (unsigned short*)alloc((size_t)D * D * 2);
    unsigned short* wk2b  = (unsigned short*)alloc((size_t)D * D * 2);
    unsigned short* wv2b  = (unsigned short*)alloc((size_t)D * D * 2);
    unsigned short* wff1b = (unsigned short*)alloc((size_t)D * DFF * 2);
    unsigned short* wff2b = (unsigned short*)alloc((size_t)DFF * D * 2);
    unsigned short* Qbf   = (unsigned short*)alloc(BS * D * 2);
    unsigned short* Kbf   = (unsigned short*)alloc(BS * D * 2);
    unsigned short* Vbf   = (unsigned short*)alloc(BS * D * 2);
    float*          scF   = (float*)alloc((size_t)B * S * S * 4);   // reused
    unsigned short* Pbf   = (unsigned short*)alloc((size_t)B * S * S * 2);
    float*          attnF = (float*)alloc(BS * D * 4);              // reused
    float*          y1F   = (float*)alloc(BS * D * 4);
    unsigned short* y1bf  = (unsigned short*)alloc(BS * D * 2);
    float*          y2F   = (float*)alloc(BS * D * 4);
    unsigned short* y2bf  = (unsigned short*)alloc(BS * D * 2);
    unsigned short* hbf   = (unsigned short*)alloc(BS * DFF * 2);
    float*          ffnF  = (float*)alloc(BS * D * 4);

    auto cast = [&](const float* src, unsigned short* dst, long long n) {
        long long n4 = n >> 2;
        int blocks = (int)((n4 + 255) / 256);
        if (blocks > 16384) blocks = 16384;
        hipLaunchKernelGGL(cast_f32_bf16_kernel, dim3(blocks), dim3(256), 0,
                           stream, src, dst, n4);
    };

    cast(y,    ybf,   BS * D);
    cast(Z,    zbf,   BS * D);
    cast(WQ1,  wq1b,  (long long)D * D);
    cast(WK1,  wk1b,  (long long)D * D);
    cast(WV1,  wv1b,  (long long)D * D);
    cast(WQ2,  wq2b,  (long long)D * D);
    cast(WK2,  wk2b,  (long long)D * D);
    cast(WV2,  wv2b,  (long long)D * D);
    cast(Wff1, wff1b, (long long)D * DFF);
    cast(Wff2, wff2b, (long long)DFF * D);

    const float qscale = 0.03125f;   // 1/sqrt(1024)
    const dim3 thr(256);
    const long long SD = (long long)S * D, SS = (long long)S * S;

    // ================= self-attention (causal) =================
    hipLaunchKernelGGL((gemm_bf16_wmma_kernel<false, true>), dim3(D/128, BS/128, 1), thr, 0, stream,
                       ybf, wq1b, (void*)Qbf, nullptr, (int)BS, D, D, 0, 0, 0, 1.0f, 0, 0);
    hipLaunchKernelGGL((gemm_bf16_wmma_kernel<false, true>), dim3(D/128, BS/128, 1), thr, 0, stream,
                       ybf, wk1b, (void*)Kbf, nullptr, (int)BS, D, D, 0, 0, 0, 1.0f, 0, 0);
    hipLaunchKernelGGL((gemm_bf16_wmma_kernel<false, true>), dim3(D/128, BS/128, 1), thr, 0, stream,
                       ybf, wv1b, (void*)Vbf, nullptr, (int)BS, D, D, 0, 0, 0, 1.0f, 0, 0);
    hipLaunchKernelGGL((gemm_bf16_wmma_kernel<true, false>), dim3(S/128, S/128, B), thr, 0, stream,
                       Qbf, Kbf, (void*)scF, nullptr, S, S, D, SD, SD, SS, qscale, 0, 1);
    hipLaunchKernelGGL(softmax_rows_kernel, dim3((unsigned)(B * S)), thr, 0, stream, scF, Pbf, S);
    hipLaunchKernelGGL((gemm_bf16_wmma_kernel<false, false>), dim3(D/128, S/128, B), thr, 0, stream,
                       Pbf, Vbf, (void*)attnF, nullptr, S, D, S, SS, SD, SD, 1.0f, 0, 0);
    hipLaunchKernelGGL(add_layernorm_kernel, dim3((unsigned)BS), thr, 0, stream,
                       y, attnF, g1, be1, y1F, y1bf, D);

    // ================= cross-attention =================
    hipLaunchKernelGGL((gemm_bf16_wmma_kernel<false, true>), dim3(D/128, BS/128, 1), thr, 0, stream,
                       y1bf, wq2b, (void*)Qbf, nullptr, (int)BS, D, D, 0, 0, 0, 1.0f, 0, 0);
    hipLaunchKernelGGL((gemm_bf16_wmma_kernel<false, true>), dim3(D/128, BS/128, 1), thr, 0, stream,
                       zbf, wk2b, (void*)Kbf, nullptr, (int)BS, D, D, 0, 0, 0, 1.0f, 0, 0);
    hipLaunchKernelGGL((gemm_bf16_wmma_kernel<false, true>), dim3(D/128, BS/128, 1), thr, 0, stream,
                       zbf, wv2b, (void*)Vbf, nullptr, (int)BS, D, D, 0, 0, 0, 1.0f, 0, 0);
    hipLaunchKernelGGL((gemm_bf16_wmma_kernel<true, false>), dim3(S/128, S/128, B), thr, 0, stream,
                       Qbf, Kbf, (void*)scF, nullptr, S, S, D, SD, SD, SS, qscale, 0, 0);
    hipLaunchKernelGGL(softmax_rows_kernel, dim3((unsigned)(B * S)), thr, 0, stream, scF, Pbf, S);
    hipLaunchKernelGGL((gemm_bf16_wmma_kernel<false, false>), dim3(D/128, S/128, B), thr, 0, stream,
                       Pbf, Vbf, (void*)attnF, nullptr, S, D, S, SS, SD, SD, 1.0f, 0, 0);
    hipLaunchKernelGGL(add_layernorm_kernel, dim3((unsigned)BS), thr, 0, stream,
                       y1F, attnF, g2, be2, y2F, y2bf, D);

    // ================= FFN =================
    hipLaunchKernelGGL((gemm_bf16_wmma_kernel<false, true>), dim3(DFF/128, BS/128, 1), thr, 0, stream,
                       y2bf, wff1b, (void*)hbf, bff1, (int)BS, DFF, D, 0, 0, 0, 1.0f, 1, 0);
    hipLaunchKernelGGL((gemm_bf16_wmma_kernel<false, false>), dim3(D/128, BS/128, 1), thr, 0, stream,
                       hbf, wff2b, (void*)ffnF, bff2, (int)BS, D, DFF, 0, 0, 0, 1.0f, 0, 0);
    hipLaunchKernelGGL(add_layernorm_kernel, dim3((unsigned)BS), thr, 0, stream,
                       y2F, ffnF, g3, be3, (float*)d_out, (unsigned short*)nullptr, D);
}
